// CoarseFineNetwork_11922829213822
// MI455X (gfx1250) — matrix operand order repaired and checked by reading, not compiled, not run on hardware
//
#include <hip/hip_runtime.h>

typedef __attribute__((ext_vector_type(16))) _Float16 v16h;
typedef __attribute__((ext_vector_type(8)))  float    v8f;

#define THREADS 256

// ---------------- elementwise / graph kernels ----------------

__global__ void zero_f32(float* __restrict__ p, long n) {
    long i = (long)blockIdx.x * blockDim.x + threadIdx.x;
    if (i < n) p[i] = 0.0f;
}

// h0[n, 0:32] = {x[n], 0, 0, ...}
__global__ void pad_x(const float* __restrict__ x, float* __restrict__ h0, int N) {
    int i = blockIdx.x * blockDim.x + threadIdx.x;
    if (i < N * 32) {
        int c = i & 31;
        h0[i] = (c == 0) ? x[i >> 5] : 0.0f;
    }
}

__global__ void deg_accum(const int* __restrict__ dst, const float* __restrict__ w,
                          float* __restrict__ deg, int E) {
    int e = blockIdx.x * blockDim.x + threadIdx.x;
    if (e < E) atomicAdd(&deg[dst[e]], w[e]);
}

__global__ void deg_to_dis(float* __restrict__ deg, int n) {
    int i = blockIdx.x * blockDim.x + threadIdx.x;
    if (i < n) {
        float d = deg[i];
        deg[i] = (d > 0.0f) ? rsqrtf(d) : 0.0f;   // deg buffer becomes dis
    }
}

__global__ void edge_norm(const int* __restrict__ src, const int* __restrict__ dst,
                          const float* __restrict__ w, const float* __restrict__ dis,
                          float* __restrict__ norm, int E) {
    int e = blockIdx.x * blockDim.x + threadIdx.x;
    if (e < E) norm[e] = dis[src[e]] * w[e] * dis[dst[e]];
}

// scatter-add propagation: xout[dst] += norm * xin[src].
// 2D grid: x = edges, y = channel group of 4. Buffers have channel stride Cpad.
__global__ void propagate(const float* __restrict__ xin, float* __restrict__ xout,
                          const int* __restrict__ src, const int* __restrict__ dst,
                          const float* __restrict__ norm, int E, int C, int Cpad) {
    int e = blockIdx.x * blockDim.x + threadIdx.x;
    if (e >= E) return;
    int c0 = blockIdx.y << 2;
    float nw = norm[e];
    const float* xs = xin + (long)src[e] * Cpad + c0;
    float*       xd = xout + (long)dst[e] * Cpad + c0;
    if (c0 + 4 <= C) {
        float4 v = *(const float4*)xs;          // b128 gather
        atomicAdd(&xd[0], nw * v.x);
        atomicAdd(&xd[1], nw * v.y);
        atomicAdd(&xd[2], nw * v.z);
        atomicAdd(&xd[3], nw * v.w);
    } else {
#pragma unroll
        for (int j = 0; j < 4; ++j)
            if (c0 + j < C) atomicAdd(&xd[j], nw * xs[j]);
    }
}

// final: out[n] = sigmoid(acc[n*Cpad + 0])
__global__ void sigmoid_k(const float* __restrict__ in, float* __restrict__ out,
                          int n, int Cpad) {
    int i = blockIdx.x * blockDim.x + threadIdx.x;
    if (i < n) out[i] = 1.0f / (1.0f + __expf(-in[(long)i * Cpad]));
}

// ---------------- weight packing ----------------
// Pack W[hop][Cin,Cout] (f32) into f16 WMMA B-fragment order:
//   Bp[ ((hop*nt + tn)*nc + ck)*512 + lane*16 + j ]
// Per lane the 16 halves are K values in table order
// (K = ck*32 + (lane<16 ? 0 : 8) + j + (j>=8 ? 8 : 0)) for column tn*16 + (lane&15).
__global__ void pack_weights(const float* __restrict__ W, _Float16* __restrict__ Bp,
                             int Cin, int Cout, int nt, int nc, int hops) {
    int t = blockIdx.x * blockDim.x + threadIdx.x;
    if (t >= hops * nt * nc * 32) return;
    int lane = t & 31;
    int q    = t >> 5;
    int ck   = q % nc;  q /= nc;
    int tn   = q % nt;
    int hop  = q / nt;
    int col   = tn * 16 + (lane & 15);
    int kbase = ck * 32 + ((lane < 16) ? 0 : 8);
    const float* Wh = W + (long)hop * Cin * Cout;
    _Float16* out = Bp + (long)t * 16;
#pragma unroll
    for (int j = 0; j < 16; ++j) {
        int kk = kbase + j + ((j >= 8) ? 8 : 0);
        float f = (kk < Cin && col < Cout) ? Wh[(long)kk * Cout + col] : 0.0f;
        out[j] = (_Float16)f;
    }
}

// ---------------- WMMA GEMM ----------------
// One wave owns a 16-row stripe and ALL NT column tiles: NT accumulators live
// in VGPRs, the A fragment is loaded once per K-chunk and reused NT times, and
// consecutive WMMAs hit independent accumulators (no D->C RAW chain).
// CinPad = NC*32, CoutPad = NT*16, N%16==0 -> zero bounds checks in hot loop.
// mode: 0 = init (+bias), 1 = accumulate, 2 = accumulate + ReLU
template <int NC, int NT>
__global__ void gemm_wmma(const float* __restrict__ A, const _Float16* __restrict__ Bp,
                          const float* __restrict__ bias, float* __restrict__ Cacc,
                          int rows16, int Cout, int mode) {
    constexpr int CinPad  = NC * 32;
    constexpr int CoutPad = NT * 16;
    int lane = threadIdx.x & 31;
    int wave = threadIdx.x >> 5;
    int tm = blockIdx.x * (blockDim.x >> 5) + wave;   // wave-uniform
    if (tm >= rows16) return;                         // whole-wave exit
    int row0 = tm << 4;

    // A layout: lanes 0-15 -> M=lane, K runs {0..7,16..23}; lanes 16-31 -> +8
    int arow   = row0 + (lane & 15);
    int kbaseA = (lane < 16) ? 0 : 8;
    const float*    ap = A + (long)arow * CinPad + kbaseA;
    const _Float16* bp = Bp + lane * 16;

    v8f acc[NT];
#pragma unroll
    for (int t = 0; t < NT; ++t) acc[t] = (v8f){};

#pragma unroll
    for (int ck = 0; ck < NC; ++ck) {
        const float* a0p = ap + ck * 32;
        float4 a0 = *(const float4*)(a0p);
        float4 a1 = *(const float4*)(a0p + 4);
        float4 a2 = *(const float4*)(a0p + 16);
        float4 a3 = *(const float4*)(a0p + 20);
        v16h a;
        a[0]  = (_Float16)a0.x;  a[1]  = (_Float16)a0.y;
        a[2]  = (_Float16)a0.z;  a[3]  = (_Float16)a0.w;
        a[4]  = (_Float16)a1.x;  a[5]  = (_Float16)a1.y;
        a[6]  = (_Float16)a1.z;  a[7]  = (_Float16)a1.w;
        a[8]  = (_Float16)a2.x;  a[9]  = (_Float16)a2.y;
        a[10] = (_Float16)a2.z;  a[11] = (_Float16)a2.w;
        a[12] = (_Float16)a3.x;  a[13] = (_Float16)a3.y;
        a[14] = (_Float16)a3.z;  a[15] = (_Float16)a3.w;
#pragma unroll
        for (int tn = 0; tn < NT; ++tn) {
            v16h b = *(const v16h*)(bp + ((long)tn * NC + ck) * 512);
            acc[tn] = __builtin_amdgcn_wmma_f32_16x16x32_f16(
                false, a, false, b, (short)0, acc[tn], false, false);
        }
    }

    // D layout: lanes 0-15 -> M=v, N=lane; lanes 16-31 -> M=v+8, N=lane-16
    int n     = lane & 15;
    int mbase = (lane >= 16) ? 8 : 0;
    long rbase = (long)(row0 + mbase) * CoutPad + n;
#pragma unroll
    for (int tn = 0; tn < NT; ++tn) {
        long base = rbase + tn * 16;
        if (mode == 1) {
#pragma unroll
            for (int v = 0; v < 8; ++v)
                Cacc[base + (long)v * CoutPad] += acc[tn][v];
        } else if (mode == 2) {
#pragma unroll
            for (int v = 0; v < 8; ++v) {
                long idx = base + (long)v * CoutPad;
                Cacc[idx] = fmaxf(Cacc[idx] + acc[tn][v], 0.0f);
            }
        } else {
            int col = tn * 16 + n;
            float bv = (col < Cout) ? bias[col] : 0.0f;
#pragma unroll
            for (int v = 0; v < 8; ++v)
                Cacc[base + (long)v * CoutPad] = acc[tn][v] + bv;
        }
    }
}

// ---------------- host orchestration ----------------

static inline int cdiv(long a, long b) { return (int)((a + b - 1) / b); }
static inline int pad32(int c) { return (c + 31) & ~31; }

typedef void (*gemm_fn)(const float*, const _Float16*, const float*, float*,
                        int, int, int);

static void launch_gemm(int layer, int gblk, hipStream_t stream,
                        const float* A, const _Float16* Bp, const float* bias,
                        float* Cacc, int rows16, int Cout, int mode) {
    switch (layer) {   // (NC, NT) per layer: CinPad/32, CoutPad/16
    case 0: gemm_wmma<1, 4><<<gblk, THREADS, 0, stream>>>(A, Bp, bias, Cacc, rows16, Cout, mode); break;
    case 1: gemm_wmma<2, 8><<<gblk, THREADS, 0, stream>>>(A, Bp, bias, Cacc, rows16, Cout, mode); break;
    case 2: gemm_wmma<4, 14><<<gblk, THREADS, 0, stream>>>(A, Bp, bias, Cacc, rows16, Cout, mode); break;
    case 3: gemm_wmma<7, 6><<<gblk, THREADS, 0, stream>>>(A, Bp, bias, Cacc, rows16, Cout, mode); break;
    case 4: gemm_wmma<3, 2><<<gblk, THREADS, 0, stream>>>(A, Bp, bias, Cacc, rows16, Cout, mode); break;
    }
}

extern "C" void kernel_launch(void* const* d_in, const int* in_sizes, int n_in,
                              void* d_out, int out_size, void* d_ws, size_t ws_size,
                              hipStream_t stream) {
    const float* x  = (const float*)d_in[0];
    const int*   ei = (const int*)d_in[1];
    const float* ew = (const float*)d_in[2];

    const int N = in_sizes[0];
    const int E = in_sizes[2];
    const int K = 20;
    const int HOPS = K + 1;
    const int DIMS[6] = {1, 60, 100, 200, 80, 1};
    const int PMAX = 224;   // pad32(200)

    const float* Wl[5];
    const float* bl[5];
    for (int l = 0; l < 5; ++l) {
        Wl[l] = (const float*)d_in[3 + 2 * l];
        bl[l] = (const float*)d_in[4 + 2 * l];
    }

    // ---- workspace layout ----
    float* deg  = (float*)d_ws;                 // N   (becomes dis)
    float* norm = deg + N;                      // E
    float* Bx   = norm + E;                     // N*32   padded layer-1 input
    float* B0   = Bx + (long)N * 32;            // N*PMAX (acc / activations)
    float* B1   = B0 + (long)N * PMAX;          // N*PMAX (propagation ping)
    float* B2   = B1 + (long)N * PMAX;          // N*PMAX (propagation pong)
    float* B3   = B2 + (long)N * PMAX;          // N*PMAX (acc / activations)
    _Float16* Bpack = (_Float16*)(B3 + (long)N * PMAX);  // packed weights
    (void)ws_size; (void)n_in; (void)out_size;

    const int* src = ei;        // edge_index[0]
    const int* dst = ei + E;    // edge_index[1]

    // ---- normalization: deg -> dis -> per-edge norm ----
    zero_f32<<<cdiv(N, THREADS), THREADS, 0, stream>>>(deg, N);
    deg_accum<<<cdiv(E, THREADS), THREADS, 0, stream>>>(dst, ew, deg, E);
    deg_to_dis<<<cdiv(N, THREADS), THREADS, 0, stream>>>(deg, N);
    edge_norm<<<cdiv(E, THREADS), THREADS, 0, stream>>>(src, dst, ew, deg, norm, E);

    // ---- pack all weights into f16 WMMA fragment order ----
    _Float16* Bp_l[5];
    {
        long off = 0;
        for (int l = 0; l < 5; ++l) {
            int Cin = DIMS[l], Cout = DIMS[l + 1];
            int nc = pad32(Cin) / 32, nt = pad32(Cout) / 16;
            Bp_l[l] = Bpack + off;
            int total = HOPS * nt * nc * 32;
            pack_weights<<<cdiv(total, THREADS), THREADS, 0, stream>>>(
                Wl[l], Bp_l[l], Cin, Cout, nt, nc, HOPS);
            off += (long)total * 16;
        }
    }

    // ---- padded layer-1 input ----
    pad_x<<<cdiv((long)N * 32, THREADS), THREADS, 0, stream>>>(x, Bx, N);

    // ---- layers ----
    const float* hin = Bx;
    float* accs[2] = {B0, B3};
    const int rows16 = N / 16;   // N = 50000 -> 3125, exact

    for (int l = 0; l < 5; ++l) {
        int Cin  = DIMS[l];
        int Cout = DIMS[l + 1];
        int CinPad = pad32(Cin), CoutPad = pad32(Cout);
        int nc = CinPad / 32, nt = CoutPad / 16;
        float* acc = accs[l & 1];

        int gblk = cdiv(rows16, THREADS / 32);
        long bp_hop = (long)nt * nc * 512;   // halves per hop

        // hop 0: acc = hin @ W[0] + b
        launch_gemm(l, gblk, stream, hin, Bp_l[l], bl[l], acc, rows16, Cout, 0);

        int ngroups = (Cin + 3) >> 2;
        dim3 pgrid(cdiv(E, THREADS), ngroups, 1);
        const float* cur = hin;
        float* nxt = B1;
        bool fuse_relu = (l < 4);
        for (int k = 1; k <= K; ++k) {
            zero_f32<<<cdiv((long)N * CinPad, THREADS), THREADS, 0, stream>>>(
                nxt, (long)N * CinPad);
            propagate<<<pgrid, THREADS, 0, stream>>>(cur, nxt, src, dst, norm,
                                                     E, Cin, CinPad);
            int mode = (k == K && fuse_relu) ? 2 : 1;
            launch_gemm(l, gblk, stream, nxt, Bp_l[l] + (long)k * bp_hop, nullptr,
                        acc, rows16, Cout, mode);
            cur = nxt;
            nxt = (cur == B1) ? B2 : B1;
        }

        if (l < 4) {
            hin = acc;   // ReLU already fused into hop-20 GEMM
        } else {
            sigmoid_k<<<cdiv(N, THREADS), THREADS, 0, stream>>>(acc, (float*)d_out,
                                                                N, CoutPad);
        }
    }
}
